// MetaPathDecoder_41996190221028
// MI455X (gfx1250) — matrix operand order repaired and checked by reading, not compiled, not run on hardware
//
#include <hip/hip_runtime.h>
#include <hip/hip_bf16.h>

typedef __attribute__((ext_vector_type(16))) _Float16 v16h;
typedef __attribute__((ext_vector_type(8)))  _Float16 h8;
typedef __attribute__((ext_vector_type(8)))  float    v8f;

#define EDIM 256
#define NB   4096
#define NT   4096

__constant__ int c_path_rel[8][3] = {
  {0,1,-1},{0,2,-1},{0,3,-1},{0,4,5},{6,0,-1},{7,8,9},{10,11,9},{10,12,0}
};

__device__ __forceinline__ v8f zero8() {
  v8f z = {0.f,0.f,0.f,0.f,0.f,0.f,0.f,0.f};
  return z;
}

// Build this lane's A fragment (16x32 f16 tile, rows striped in-lane):
// lane half = l>>4 selects contiguous K chunk [ka,ka+16).
__device__ __forceinline__ v16h load_a(const _Float16* __restrict__ xr, int ka) {
  return *(const v16h*)(xr + ka);
}

// ---- generic Y = X @ W^T (+bias)(relu?) ; one wave per 16x64 output tile ----
// X, W are f16 row-major. Y is f32; Yh optional f16 mirror.
__global__ void linear_xwt_kernel(const _Float16* __restrict__ Xh, int ldx,
                                  const _Float16* __restrict__ Wh, int ldw,
                                  const float* __restrict__ bias,
                                  float* __restrict__ Y,
                                  _Float16* __restrict__ Yh, int ldy,
                                  int M, int N, int K, int relu) {
  const int lane = threadIdx.x & 31;
  const int wave = threadIdx.x >> 5;
  const int tile = blockIdx.x * (blockDim.x >> 5) + wave;
  const int ntn  = N >> 6;                    // macro col-tiles of 64
  const int ntot = (M >> 4) * ntn;
  if (tile >= ntot) return;                   // wave-uniform: EXEC stays all-ones
  const int rt = tile / ntn, cm = tile % ntn;
  const int half = lane >> 4, idx = lane & 15;
  const int r0 = rt << 4, c0 = cm << 6;
  const _Float16* xr  = Xh + (size_t)(r0 + idx) * ldx;
  const _Float16* wr0 = Wh + (size_t)(c0 + idx) * ldw;
  const size_t ws16 = (size_t)16 * ldw;
  v8f acc[4] = {zero8(), zero8(), zero8(), zero8()};
  for (int k = 0; k < K; k += 32) {
    const int ka = k + half * 16;
    const int kb = k + half * 16;
    __builtin_prefetch(xr + ka + 64, 0, 0);
    __builtin_prefetch(wr0 + kb + 64, 0, 0);
    v16h a  = load_a(xr, ka);
    v16h b0 = *(const v16h*)(wr0 + kb);
    v16h b1 = *(const v16h*)(wr0 + ws16 + kb);
    v16h b2 = *(const v16h*)(wr0 + 2 * ws16 + kb);
    v16h b3 = *(const v16h*)(wr0 + 3 * ws16 + kb);
    acc[0] = __builtin_amdgcn_wmma_f32_16x16x32_f16(false, a, false, b0, (short)0, acc[0], false, false);
    acc[1] = __builtin_amdgcn_wmma_f32_16x16x32_f16(false, a, false, b1, (short)0, acc[1], false, false);
    acc[2] = __builtin_amdgcn_wmma_f32_16x16x32_f16(false, a, false, b2, (short)0, acc[2], false, false);
    acc[3] = __builtin_amdgcn_wmma_f32_16x16x32_f16(false, a, false, b3, (short)0, acc[3], false, false);
  }
#pragma unroll
  for (int t = 0; t < 4; ++t) {
    const int col = c0 + 16 * t + idx;
    const float bv = bias ? bias[col] : 0.f;
#pragma unroll
    for (int j = 0; j < 8; ++j) {
      float v = acc[t][j] + bv;
      if (relu) v = fmaxf(v, 0.f);
      const size_t o = (size_t)(r0 + j + 8 * half) * ldy + col;
      Y[o] = v;
      if (Yh) Yh[o] = (_Float16)v;
    }
  }
}

// ---- gathers: Xcat (f16), Temb (f16), Remb (f32) ----
__global__ void gather_kernel(const float* __restrict__ E, const float* __restrict__ R,
                              const int* __restrict__ h_idx, const int* __restrict__ cat_idx,
                              const int* __restrict__ t_idx, const int* __restrict__ r_idx,
                              _Float16* __restrict__ XcatH, _Float16* __restrict__ TembH,
                              float* __restrict__ Remb) {
  const int t = blockIdx.x * blockDim.x + threadIdx.x;
  const int n1 = NB * 512, n2 = NT * EDIM, n3 = NB * EDIM;
  if (t < n1) {
    int b = t >> 9, d = t & 511;
    float v = (d < 256) ? E[(size_t)h_idx[b] * EDIM + d]
                        : E[(size_t)cat_idx[b] * EDIM + (d - 256)];
    XcatH[t] = (_Float16)v;
  } else if (t < n1 + n2) {
    int u = t - n1, b = u >> 8, d = u & 255;
    TembH[u] = (_Float16)E[(size_t)t_idx[b] * EDIM + d];
  } else if (t < n1 + n2 + n3) {
    int u = t - n1 - n2, b = u >> 8, d = u & 255;
    Remb[u] = R[(size_t)r_idx[b] * EDIM + d];
  }
}

__global__ void f32_to_f16_kernel(const float* __restrict__ src,
                                  _Float16* __restrict__ dst, int n) {
  const int t = blockIdx.x * blockDim.x + threadIdx.x;
  if (t < n) dst[t] = (_Float16)src[t];
}

// ---- gi[i][s][g] = R[rel] . w_ih[i][g][:] + b_ih[i][g] ----
__global__ void gi_kernel(const float* __restrict__ R, const float* __restrict__ w_ih,
                          const float* __restrict__ b_ih, float* __restrict__ gi) {
  const int t = blockIdx.x * blockDim.x + threadIdx.x;
  if (t >= 8 * 3 * 768) return;
  const int g = t % 768, ps = t / 768, i = ps / 3, s = ps % 3;
  const int rel = c_path_rel[i][s];
  if (rel < 0) { gi[t] = 0.f; return; }
  const float* rv = R + (size_t)rel * EDIM;
  const float* wp = w_ih + ((size_t)i * 768 + g) * EDIM;
  float acc = b_ih[i * 768 + g];
  for (int k = 0; k < EDIM; ++k) acc += rv[k] * wp[k];
  gi[t] = acc;
}

// ---- fused GRU step (f16 state): gh = h @ w_hh^T + b_hh ; gates ; h' ----
// A fragment loaded once per k-step, feeds all three gate WMMAs.
__global__ void gru_step_kernel(const _Float16* __restrict__ Hsrc,
                                const _Float16* __restrict__ Whh,  // [768][256] f16
                                const float* __restrict__ bhh,     // [768]
                                const float* __restrict__ gi,      // [768] (incl b_ih)
                                _Float16* __restrict__ Hdst) {
  const int lane = threadIdx.x & 31;
  const int wave = threadIdx.x >> 5;
  const int tile = blockIdx.x * (blockDim.x >> 5) + wave;   // 4096 tiles: 256r x 16c
  const int rt = tile >> 4, ct = tile & 15;
  const int half = lane >> 4, idx = lane & 15;
  const int r0 = rt << 4, c0 = ct << 4;
  const _Float16* xr  = Hsrc + (size_t)(r0 + idx) * EDIM;
  const _Float16* wrr = Whh + (size_t)(0 * EDIM + c0 + idx) * EDIM;
  const _Float16* wrz = Whh + (size_t)(1 * EDIM + c0 + idx) * EDIM;
  const _Float16* wrn = Whh + (size_t)(2 * EDIM + c0 + idx) * EDIM;
  v8f ar = zero8(), az = zero8(), an = zero8();
  for (int k = 0; k < EDIM; k += 32) {
    const int ka = k + half * 16;
    const int kb = k + half * 16;
    v16h a  = load_a(xr, ka);
    v16h br = *(const v16h*)(wrr + kb);
    v16h bz = *(const v16h*)(wrz + kb);
    v16h bn = *(const v16h*)(wrn + kb);
    ar = __builtin_amdgcn_wmma_f32_16x16x32_f16(false, a, false, br, (short)0, ar, false, false);
    az = __builtin_amdgcn_wmma_f32_16x16x32_f16(false, a, false, bz, (short)0, az, false, false);
    an = __builtin_amdgcn_wmma_f32_16x16x32_f16(false, a, false, bn, (short)0, an, false, false);
  }
  const int col = c0 + idx;
  const float gir = gi[col],            bir  = bhh[col];
  const float giz = gi[EDIM + col],     biz  = bhh[EDIM + col];
  const float gin = gi[2 * EDIM + col], bin_ = bhh[2 * EDIM + col];
#pragma unroll
  for (int j = 0; j < 8; ++j) {
    const int row = r0 + j + 8 * half;
    const float h = (float)Hsrc[(size_t)row * EDIM + col];
    const float r = 1.f / (1.f + __expf(-(gir + ar[j] + bir)));
    const float z = 1.f / (1.f + __expf(-(giz + az[j] + biz)));
    const float n = tanhf(gin + r * (an[j] + bin_));
    Hdst[(size_t)row * EDIM + col] = (_Float16)((1.f - z) * n + z * h);
  }
}

// ---- scores + softmax + attn_out(f16) ; one block per batch row ----
__global__ void attn_kernel(const float* __restrict__ Q, const float* __restrict__ Kb,
                            const float* __restrict__ Vb, _Float16* __restrict__ AOh,
                            float* __restrict__ attnW) {
  const int b = blockIdx.x;
  const int tid = threadIdx.x, w = tid >> 5, lane = tid & 31;
  __shared__ float sc[8];
  const float* q = Q + (size_t)b * EDIM;
  const float* k = Kb + ((size_t)w * NB + b) * EDIM;
  float dot = 0.f;
#pragma unroll
  for (int j = 0; j < 8; ++j) dot += q[lane * 8 + j] * k[lane * 8 + j];
  for (int m = 16; m; m >>= 1) dot += __shfl_xor(dot, m, 32);
  if (lane == 0) sc[w] = dot * (1.f / 16.f);   // / sqrt(256)
  __syncthreads();
  float mx = sc[0];
#pragma unroll
  for (int s = 1; s < 8; ++s) mx = fmaxf(mx, sc[s]);
  float e[8], sum = 0.f;
#pragma unroll
  for (int s = 0; s < 8; ++s) { e[s] = __expf(sc[s] - mx); sum += e[s]; }
  const float inv = 1.f / sum;
  if (tid < 8) attnW[(size_t)b * 8 + tid] = e[tid] * inv;
  float acc = 0.f;
#pragma unroll
  for (int s = 0; s < 8; ++s)
    acc += e[s] * inv * Vb[((size_t)s * NB + b) * EDIM + tid];
  AOh[(size_t)b * EDIM + tid] = (_Float16)acc;
}

__global__ void hr_kernel(const float* __restrict__ H, const float* __restrict__ R,
                          _Float16* __restrict__ HRh) {
  const int t = blockIdx.x * blockDim.x + threadIdx.x;
  if (t < NB * EDIM) HRh[t] = (_Float16)(H[t] * R[t]);
}

// row L2 norms of pred1 (blocks 0..4095) and pred2 (4096..8191)
__global__ void rownorm_kernel(const float* __restrict__ P1, const float* __restrict__ P2,
                               float* __restrict__ norms) {
  const int rb = blockIdx.x;
  const float* p = ((rb < NB) ? P1 : P2) + (size_t)(rb & (NB - 1)) * NT;
  float s = 0.f;
  for (int j = threadIdx.x; j < NT; j += 256) { float v = p[j]; s += v * v; }
  __shared__ float red[256];
  red[threadIdx.x] = s; __syncthreads();
  for (int st = 128; st; st >>= 1) {
    if (threadIdx.x < st) red[threadIdx.x] += red[threadIdx.x + st];
    __syncthreads();
  }
  if (threadIdx.x == 0) norms[rb] = sqrtf(red[0]);
}

__global__ void combine_kernel(float* __restrict__ P1, const float* __restrict__ P2,
                               const float* __restrict__ norms) {
  const size_t t = (size_t)blockIdx.x * 256 + threadIdx.x;
  if (t >= (size_t)NB * NT) return;
  const int b = (int)(t >> 12);
  const float n1 = fmaxf(norms[b], 1e-12f);
  const float n2 = fmaxf(norms[NB + b], 1e-12f);
  P1[t] = 0.5f * P1[t] / n1 + 0.5f * P2[t] / n2;
}

extern "C" void kernel_launch(void* const* d_in, const int* in_sizes, int n_in,
                              void* d_out, int out_size, void* d_ws, size_t ws_size,
                              hipStream_t stream) {
  const float* E     = (const float*)d_in[0];
  const float* R     = (const float*)d_in[1];
  const float* Wmlp  = (const float*)d_in[2];
  const float* bmlp  = (const float*)d_in[3];
  const float* gWih  = (const float*)d_in[4];
  const float* gWhh  = (const float*)d_in[5];
  const float* gBih  = (const float*)d_in[6];
  const float* gBhh  = (const float*)d_in[7];
  const float* Wq    = (const float*)d_in[8];
  const float* bq    = (const float*)d_in[9];
  const float* Wk    = (const float*)d_in[10];
  const float* bk    = (const float*)d_in[11];
  const float* Wv    = (const float*)d_in[12];
  const float* bv    = (const float*)d_in[13];
  const float* Wo    = (const float*)d_in[14];
  const float* bo    = (const float*)d_in[15];
  const int* h_idx   = (const int*)d_in[16];
  const int* r_idx   = (const int*)d_in[17];
  const int* t_idx   = (const int*)d_in[18];
  const int* cat_idx = (const int*)d_in[19];
  float* OUT   = (float*)d_out;                       // pred: 4096*4096
  float* attnW = OUT + (size_t)NB * NT;               // attn : 4096*8

  // workspace layout (bytes)
  char* base = (char*)d_ws;
  size_t off = 0;
  auto allocF = [&](size_t n) { float* p = (float*)(base + off); off += n * 4; return p; };
  auto allocH = [&](size_t n) { _Float16* p = (_Float16*)(base + off); off += n * 2; return p; };
  const size_t PS = (size_t)NB * EDIM;                // 1,048,576 elements
  // f32 regions
  float* Remb = allocF(PS);
  float* Hmlp = allocF(PS);
  float* Kbuf = allocF(8 * PS);        // pred2 overlays Kbuf+Vbuf (exactly 16M floats)
  float* Vbuf = allocF(8 * PS);
  float* Q    = allocF(PS);
  float* AWo  = allocF(PS);            // f32 copy of AWo (unused downstream; mirror is)
  float* GI   = allocF(8 * 3 * 768);
  float* NRM  = allocF(2 * NB);
  // f16 regions
  _Float16* XcatH = allocH((size_t)NB * 512);
  _Float16* TembH = allocH(PS);
  _Float16* HmlpH = allocH(PS);
  _Float16* HAh   = allocH(8 * PS);
  _Float16* HBh   = allocH(8 * PS);
  _Float16* AOh   = allocH(PS);
  _Float16* AWoH  = allocH(PS);
  _Float16* HRh   = allocH(PS);
  _Float16* WmlpH = allocH((size_t)EDIM * 512);
  _Float16* WqH   = allocH((size_t)EDIM * EDIM);
  _Float16* WkH   = allocH((size_t)EDIM * EDIM);
  _Float16* WvH   = allocH((size_t)EDIM * EDIM);
  _Float16* WoH   = allocH((size_t)EDIM * EDIM);
  _Float16* WhhH  = allocH((size_t)8 * 768 * EDIM);
  float* P2 = Kbuf;   // K/V dead after attn; 8M+8M floats == 4096*4096 exactly

  (void)Vbuf;

  auto linear = [&](const _Float16* X, int ldx, const _Float16* W_, int ldw,
                    const float* bias, float* Y, _Float16* Yh, int ldy,
                    int M, int N, int K, int relu) {
    int tiles = (M >> 4) * (N >> 6);
    linear_xwt_kernel<<<(tiles + 7) / 8, 256, 0, stream>>>(X, ldx, W_, ldw, bias,
                                                           Y, Yh, ldy, M, N, K, relu);
  };
  auto cvt = [&](const float* s, _Float16* d, int n) {
    f32_to_f16_kernel<<<(n + 255) / 256, 256, 0, stream>>>(s, d, n);
  };

  // 1) gathers (f16 Xcat/Temb, f32 Remb)
  {
    int tot = NB * 512 + NT * EDIM + NB * EDIM;
    gather_kernel<<<(tot + 255) / 256, 256, 0, stream>>>(E, R, h_idx, cat_idx,
                                                         t_idx, r_idx, XcatH, TembH, Remb);
  }
  // 2) weight conversions + broadcast GRU input gates
  cvt(Wmlp, WmlpH, EDIM * 512);
  cvt(Wq, WqH, EDIM * EDIM);
  cvt(Wk, WkH, EDIM * EDIM);
  cvt(Wv, WvH, EDIM * EDIM);
  cvt(Wo, WoH, EDIM * EDIM);
  cvt(gWhh, WhhH, 8 * 768 * EDIM);
  gi_kernel<<<(8 * 3 * 768 + 255) / 256, 256, 0, stream>>>(R, gWih, gBih, GI);
  // 3) MLP: Hmlp = relu(Xcat @ Wmlp^T + bmlp)  (f32 + f16 mirror)
  linear(XcatH, 512, WmlpH, 512, bmlp, Hmlp, HmlpH, EDIM, NB, EDIM, 512, 1);
  // 4) GRU recurrences (20 sequenced steps, f16 state)
  static const int path_len_h[8] = {2, 2, 2, 3, 2, 3, 3, 3};
  const _Float16* finalsH[8];
  for (int i = 0; i < 8; ++i) {
    _Float16* bufA = HAh + (size_t)i * PS;
    _Float16* bufB = HBh + (size_t)i * PS;
    const _Float16* src = HmlpH;
    for (int s = 0; s < path_len_h[i]; ++s) {
      _Float16* dst = (s & 1) ? bufB : bufA;
      gru_step_kernel<<<512, 256, 0, stream>>>(src, WhhH + (size_t)i * 768 * EDIM,
                                               gBhh + (size_t)i * 768,
                                               GI + ((size_t)i * 3 + s) * 768, dst);
      src = dst;
    }
    finalsH[i] = src;
  }
  // 5) q / k / v
  linear(HmlpH, EDIM, WqH, EDIM, bq, Q, nullptr, EDIM, NB, EDIM, EDIM, 0);
  for (int i = 0; i < 8; ++i) {
    linear(finalsH[i], EDIM, WkH, EDIM, bk, Kbuf + (size_t)i * PS, nullptr, EDIM, NB, EDIM, EDIM, 0);
    linear(finalsH[i], EDIM, WvH, EDIM, bv, Vbuf + (size_t)i * PS, nullptr, EDIM, NB, EDIM, EDIM, 0);
  }
  // 6) attention (scores, softmax, weighted sum) + export attn weights
  attn_kernel<<<NB, 256, 0, stream>>>(Q, Kbuf, Vbuf, AOh, attnW);
  // 7) output projection
  linear(AOh, EDIM, WoH, EDIM, bo, AWo, AWoH, EDIM, NB, EDIM, EDIM, 0);
  // 8) pred1 = AWo @ Temb^T  (into d_out, pre-normalization)
  linear(AWoH, EDIM, TembH, EDIM, nullptr, OUT, nullptr, NT, NB, NT, EDIM, 0);
  // 9) pred2 = (Hmlp * Remb) @ Temb^T   (P2 overlays dead K/V)
  hr_kernel<<<((int)PS + 255) / 256, 256, 0, stream>>>(Hmlp, Remb, HRh);
  linear(HRh, EDIM, TembH, EDIM, nullptr, P2, nullptr, NT, NB, NT, EDIM, 0);
  // 10) row norms + combine
  rownorm_kernel<<<2 * NB, 256, 0, stream>>>(OUT, P2, NRM);
  combine_kernel<<<(int)(((size_t)NB * NT + 255) / 256), 256, 0, stream>>>(OUT, P2, NRM);
  (void)in_sizes; (void)n_in; (void)out_size; (void)ws_size;
}
